// GCN_32839319945453
// MI455X (gfx1250) — matrix operand order repaired and checked by reading, not compiled, not run on hardware
//
#include <hip/hip_runtime.h>
#include <hip/hip_bf16.h>

// GCN on MI455X (gfx1250), wave32.
//   h    = emb_table                         (x == arange)
//   agg  = SpMM(adj, h)                      f32 hw atomics, L2-resident
//   y[i] = relu(agg@W1 + b1)[i] . W2         bf16 WMMA GEMM, fused epilogue
//   z    = SpMM(adj, y)                      scalar atomics (W2 pushed through segment_sum)
//   out  = (z + b2) * x

typedef __attribute__((ext_vector_type(16))) __bf16 v16bf;
typedef __attribute__((ext_vector_type(8)))  float  v8f;

#define N_NODES 50000
#define N_EDGES 800000
#define EMB_DIM 256
#define HID_DIM 512

__device__ __forceinline__ void atomAddF32(float* p, float v) {
    __hip_atomic_fetch_add(p, v, __ATOMIC_RELAXED, __HIP_MEMORY_SCOPE_AGENT);
}

// ---------------- init: zero agg / y / z ----------------
__global__ void k_init(float* __restrict__ agg, float* __restrict__ y, float* __restrict__ z) {
    int i = blockIdx.x * blockDim.x + threadIdx.x;          // 12.8M threads exactly
    agg[i] = 0.0f;
    if (i < N_NODES) { y[i] = 0.0f; z[i] = 0.0f; }
}

// ---------------- W1 [K=256][N=512] f32 -> W1t [N][K] bf16 ----------------
__global__ void k_w1t(const float* __restrict__ W1, __bf16* __restrict__ W1t) {
    int i = blockIdx.x * blockDim.x + threadIdx.x;          // 131072 threads
    int k = i / HID_DIM, n = i % HID_DIM;
    W1t[n * EMB_DIM + k] = (__bf16)W1[i];
}

// ---------------- SpMM1: agg[row[e]] += w[e] * emb[col[e]] ----------------
// one thread per (edge, 4-float chunk): 800000 * 64 threads
__global__ void k_spmm1(const int* __restrict__ row, const int* __restrict__ col,
                        const float* __restrict__ ew, const float* __restrict__ emb,
                        float* __restrict__ agg) {
    int t = blockIdx.x * blockDim.x + threadIdx.x;
    int e = t >> 6;
    int c = (t & 63) << 2;
    int r  = row[e];
    int cl = col[e];
    float w = ew[e];
    const float4 v = *(const float4*)(emb + cl * EMB_DIM + c);
    float* dst = agg + r * EMB_DIM + c;
    atomAddF32(dst + 0, w * v.x);
    atomAddF32(dst + 1, w * v.y);
    atomAddF32(dst + 2, w * v.z);
    atomAddF32(dst + 3, w * v.w);
}

// ---------------- A fragment: 16 contiguous f32 -> v16bf, contiguous k order (k = 16h + s) ----------------
__device__ __forceinline__ v16bf load_a_frag(const float* __restrict__ base) {
    const float4* q = (const float4*)base;
    float4 u0 = q[0], u1 = q[1];   // K offsets +0..7
    float4 u2 = q[2], u3 = q[3];   // K offsets +8..15
    v16bf r;
    r[0]  = (__bf16)u0.x; r[1]  = (__bf16)u0.y; r[2]  = (__bf16)u0.z; r[3]  = (__bf16)u0.w;
    r[4]  = (__bf16)u1.x; r[5]  = (__bf16)u1.y; r[6]  = (__bf16)u1.z; r[7]  = (__bf16)u1.w;
    r[8]  = (__bf16)u2.x; r[9]  = (__bf16)u2.y; r[10] = (__bf16)u2.z; r[11] = (__bf16)u2.w;
    r[12] = (__bf16)u3.x; r[13] = (__bf16)u3.y; r[14] = (__bf16)u3.z; r[15] = (__bf16)u3.w;
    return r;
}

// ---------------- GEMM1 + fused (bias, relu, .W2) -> y atomics ----------------
// grid (391, 4); block 256 (8 waves). Block tile 128x128, wave tile 32x64.
__global__ __launch_bounds__(256) void k_gemm_y(const float* __restrict__ agg,
                                                const __bf16* __restrict__ W1t,
                                                const float* __restrict__ b1,
                                                const float* __restrict__ W2,
                                                float* __restrict__ y) {
    const int lane = threadIdx.x & 31;
    const int wv   = threadIdx.x >> 5;
    const int wm   = wv >> 1;            // 0..3  -> 32-row sub-block
    const int wn   = wv & 1;             // 0..1  -> 64-col sub-block
    const int m0   = blockIdx.x * 128 + wm * 32;
    const int n0   = blockIdx.y * 128 + wn * 64;
    const int lh   = lane & 15;
    const bool hi  = lane >= 16;

    v8f acc[2][4];
    #pragma unroll
    for (int i = 0; i < 2; ++i)
        #pragma unroll
        for (int t = 0; t < 4; ++t) acc[i][t] = v8f{};

    int arow0 = m0 + lh;       if (arow0 > N_NODES - 1) arow0 = N_NODES - 1;
    int arow1 = m0 + 16 + lh;  if (arow1 > N_NODES - 1) arow1 = N_NODES - 1;
    const float* ap0 = agg + arow0 * EMB_DIM;
    const float* ap1 = agg + arow1 * EMB_DIM;
    const int akoff = hi ? 16 : 0;   // A: lane-half selects K 0..15 / 16..31
    const int bkoff = hi ? 16 : 0;   // B: lane-half selects K 0..15 / 16..31

    #pragma unroll
    for (int k0 = 0; k0 < EMB_DIM; k0 += 32) {
        v16bf a0 = load_a_frag(ap0 + k0 + akoff);
        v16bf a1 = load_a_frag(ap1 + k0 + akoff);
        #pragma unroll
        for (int t = 0; t < 4; ++t) {
            int ncol = n0 + 16 * t + lh;
            const v16bf b = *(const v16bf*)(W1t + ncol * EMB_DIM + k0 + bkoff);
            acc[0][t] = __builtin_amdgcn_wmma_f32_16x16x32_bf16(
                false, a0, false, b, (short)0, acc[0][t], false, false);
            acc[1][t] = __builtin_amdgcn_wmma_f32_16x16x32_bf16(
                false, a1, false, b, (short)0, acc[1][t], false, false);
        }
    }

    // epilogue: bias + relu + dot with W2, reduce over N, atomically add into y
    #pragma unroll
    for (int i = 0; i < 2; ++i) {
        float rsum[8];
        #pragma unroll
        for (int r = 0; r < 8; ++r) rsum[r] = 0.0f;
        #pragma unroll
        for (int t = 0; t < 4; ++t) {
            int ncol   = n0 + 16 * t + lh;
            float bias = b1[ncol];
            float w2   = W2[ncol];
            #pragma unroll
            for (int r = 0; r < 8; ++r) {
                float v = acc[i][t][r] + bias;
                v = v > 0.0f ? v : 0.0f;
                rsum[r] += v * w2;
            }
        }
        // lanes 0..15 hold rows base..base+7 (distinct n); lanes 16..31 rows +8
        #pragma unroll
        for (int off = 8; off >= 1; off >>= 1)
            #pragma unroll
            for (int r = 0; r < 8; ++r)
                rsum[r] += __shfl_xor(rsum[r], off, 32);
        if (lh == 0) {
            int rowbase = m0 + 16 * i + (hi ? 8 : 0);
            #pragma unroll
            for (int r = 0; r < 8; ++r) {
                int rr = rowbase + r;
                if (rr < N_NODES) atomAddF32(y + rr, rsum[r]);
            }
        }
    }
}

// ---------------- SpMM2 collapsed to scalar: z[row[e]] += w[e] * y[col[e]] ----------------
__global__ void k_spmm2(const int* __restrict__ row, const int* __restrict__ col,
                        const float* __restrict__ ew, const float* __restrict__ y,
                        float* __restrict__ z) {
    int e = blockIdx.x * blockDim.x + threadIdx.x;
    if (e >= N_EDGES) return;
    atomAddF32(z + row[e], ew[e] * y[col[e]]);
}

// ---------------- out[i] = (z[i] + b2) * x[i] ----------------
__global__ void k_final(const float* __restrict__ z, const float* __restrict__ b2,
                        const int* __restrict__ x, float* __restrict__ out) {
    int i = blockIdx.x * blockDim.x + threadIdx.x;
    if (i < N_NODES) out[i] = (z[i] + b2[0]) * (float)x[i];
}

extern "C" void kernel_launch(void* const* d_in, const int* in_sizes, int n_in,
                              void* d_out, int out_size, void* d_ws, size_t ws_size,
                              hipStream_t stream) {
    (void)in_sizes; (void)n_in; (void)out_size; (void)ws_size;
    const int*   x   = (const int*)d_in[0];
    const int*   row = (const int*)d_in[1];
    const int*   col = (const int*)d_in[2];
    const float* ew  = (const float*)d_in[3];
    const float* emb = (const float*)d_in[4];
    const float* W1  = (const float*)d_in[5];
    const float* b1  = (const float*)d_in[6];
    const float* W2  = (const float*)d_in[7];
    const float* b2  = (const float*)d_in[8];
    float* out = (float*)d_out;

    // workspace layout (agg/W1t/y offsets 256B-aligned; z only 64B)
    char*   ws  = (char*)d_ws;
    float*  agg = (float*)ws;                                  // 50000*256*4 = 51,200,000 B
    __bf16* W1t = (__bf16*)(ws + 51200000);                    // 512*256*2   =    262,144 B
    float*  y   = (float*)(ws + 51200000 + 262144);            // 200,000 B
    float*  z   = y + N_NODES;                                 // 200,000 B

    k_init <<<(N_NODES * EMB_DIM) / 256, 256, 0, stream>>>(agg, y, z);
    k_w1t  <<<(EMB_DIM * HID_DIM) / 256, 256, 0, stream>>>(W1, W1t);
    k_spmm1<<<(N_EDGES * 64) / 256, 256, 0, stream>>>(row, col, ew, emb, agg);
    dim3 g((N_NODES + 127) / 128, HID_DIM / 128);
    k_gemm_y<<<g, 256, 0, stream>>>(agg, W1t, b1, W2, y);
    k_spmm2<<<(N_EDGES + 255) / 256, 256, 0, stream>>>(row, col, ew, y, z);
    k_final<<<(N_NODES + 255) / 256, 256, 0, stream>>>(z, b2, x, out);
}
